// GRURegressor_74062416052991
// MI455X (gfx1250) — matrix operand order repaired and checked by reading, not compiled, not run on hardware
//
#include <hip/hip_runtime.h>
#include <hip/hip_bf16.h>

// ---------------------------------------------------------------------------
// GRU regressor for MI455X (gfx1250, wave32, v_wmma_f32_16x16x32_f16).
//
// One wave32 owns a 16-row batch tile persistently across T=512 steps.
//  - All 36/48 B-fragments ([W_ih | W_hh] in exact WMMA B layout) are gathered
//    straight from global memory into registers ONCE (each lane's fragment is
//    two runs of 8 consecutive K-values of one weight row -> contiguous f32 loads) and
//    stay resident for the whole time loop (~384 VGPRs; occupancy is 1
//    wave/SIMD anyway with only B/16 = 64 waves on the whole chip).
//  - Biases live in persistent v8f fragments used as the C operand of the
//    first WMMA of each accumulation chain (no per-step splat moves).
//  - n-gate keeps x- and h- contributions in separate accumulators
//    (n = tanh(xn + r*hn) needs them apart).
//  - Hidden state stays in registers (C layout); per-step transpose to A
//    layout goes through a 4KB double-buffered LDS staging area.
//  - Output sequence is written with the CDNA5 async store engine:
//    global_store_async_from_lds_b128, fenced with s_wait_dscnt /
//    s_wait_asynccnt (ASYNCcnt), double-buffered so stores overlap compute.
// ---------------------------------------------------------------------------

typedef _Float16 v16h __attribute__((ext_vector_type(16)));
typedef _Float16 v8h  __attribute__((ext_vector_type(8)));
typedef float    v8f  __attribute__((ext_vector_type(8)));
typedef __attribute__((address_space(3))) _Float16 lds_h;

#define GRU_B 1024
#define GRU_T 512
#define GRU_H 64

__device__ __forceinline__ float fast_sigmoid(float x) {
  // 1 / (1 + e^-x) with v_exp + v_rcp (no IEEE divide chain)
  return __builtin_amdgcn_rcpf(1.0f + __expf(-x));
}
__device__ __forceinline__ float fast_tanh(float x) {
  // 1 - 2/(e^{2x}+1); inf-safe: e->inf => rcp->0 => 1; e->0 => -1
  return 1.0f - 2.0f * __builtin_amdgcn_rcpf(__expf(2.0f * x) + 1.0f);
}

// Issue one async LDS->global 16B store (ASYNCcnt-tracked).
__device__ __forceinline__ void async_store_b128(const _Float16* gptr,
                                                 const _Float16* lptr) {
  unsigned long long ga = (unsigned long long)(uintptr_t)gptr;
  unsigned la = (unsigned)(unsigned long long)(lds_h*)lptr;  // LDS byte offset
  asm volatile("global_store_async_from_lds_b128 %0, %1, off"
               :: "v"(ga), "v"(la) : "memory");
}

// ---------------------------------------------------------------------------
// GRU layer kernel.
//   DIN  : true input width (16 layer0, 64 else), XPAD: padded to 32-multiple
//   XF32 : layer0 reads fp32 x, else f16 workspace
//   STORE_ALL: write [B,T,H] f16 sequence; STORE_H: write final h fp32
// A = [ x_t (pad XPAD) | h (64) ], K = XPAD + 64.
// ---------------------------------------------------------------------------
template <int DIN, int XPAD, bool XF32, bool STORE_ALL, bool STORE_H>
__global__ __launch_bounds__(32, 1) void gru_layer(
    const float* __restrict__ xf,
    const _Float16* __restrict__ xh,
    const float* __restrict__ Wih,       // [3H, DIN]
    const float* __restrict__ Whh,       // [3H, H]
    const float* __restrict__ bih,       // [3H]
    const float* __restrict__ bhh,       // [3H]
    _Float16* __restrict__ out,          // [B,T,H] f16 (STORE_ALL)
    float* __restrict__ hlast)           // [B,H] fp32 (STORE_H)
{
  constexpr int KTX   = XPAD / 32;       // K-tiles covering x
  constexpr int KT    = KTX + 2;         // + 2 K-tiles covering h
  constexpr int NFRAG = KT * 12;

  // 4KB LDS: double-buffered 16x64 f16 h staging (row-major)
  extern __shared__ char smem_raw[];
  _Float16* hbuf = (_Float16*)smem_raw;  // 2 * 1024 halves

  const int lane = threadIdx.x & 31;
  const int row  = lane & 15;            // A-row / B-column / C-column
  const int hi   = lane >> 4;
  const int b0   = blockIdx.x * 16;

  // ---- zero both h staging buffers (h0 = 0)
  {
    v8h z;
#pragma unroll
    for (int i = 0; i < 8; i++) z[i] = (_Float16)0.0f;
#pragma unroll
    for (int k = 0; k < 8; k++) *(v8h*)(hbuf + lane * 64 + k * 8) = z;
  }

  // ---- gather ALL B fragments into registers (resident across the T loop).
  // Dense 16-bit B 32x16 layout (matching A): element e<8 holds K=hi*8+e,
  // e>=8 holds K=16+hi*8+(e-8), column N = lane&15.  So per (kt,nt) this
  // lane needs W[N][kt*32 + hi*8 + e] and W[N][kt*32 + 16 + hi*8 + e-8].
  v16h bfr[NFRAG];
#pragma unroll
  for (int kt = 0; kt < KT; kt++) {
#pragma unroll
    for (int nt = 0; nt < 12; nt++) {
      const int N = nt * 16 + row;
      v16h f;
      if (kt < KTX) {
#pragma unroll
        for (int e = 0; e < 16; e++) {
          const int K = kt * 32 + (e < 8 ? hi * 8 + e : 16 + hi * 8 + (e - 8));
          f[e] = (K < DIN) ? (_Float16)Wih[(size_t)N * DIN + K]
                           : (_Float16)0.0f;
        }
      } else {
        const float* wp = Whh + (size_t)N * GRU_H + (kt - KTX) * 32 + hi * 8;
#pragma unroll
        for (int e = 0; e < 16; e++) f[e] = (_Float16)wp[e < 8 ? e : e + 8];
      }
      bfr[kt * 12 + nt] = f;
    }
  }

  // ---- persistent bias fragments (used as C operand of first WMMA in chain)
  v8f bFragRZ[8], bFragXN[4], bFragHN[4];
#pragma unroll
  for (int g = 0; g < 8; g++) {
    const float b = bih[g * 16 + row] + bhh[g * 16 + row];
#pragma unroll
    for (int i = 0; i < 8; i++) bFragRZ[g][i] = b;
  }
#pragma unroll
  for (int q = 0; q < 4; q++) {
    const float bx = bih[128 + q * 16 + row];
    const float bh = bhh[128 + q * 16 + row];
#pragma unroll
    for (int i = 0; i < 8; i++) { bFragXN[q][i] = bx; bFragHN[q][i] = bh; }
  }

  // ---- persistent hidden state, C-fragment layout
  v8f hC[4];
#pragma unroll
  for (int q = 0; q < 4; q++)
#pragma unroll
    for (int i = 0; i < 8; i++) hC[q][i] = 0.0f;

  // =========================== time loop ===================================
  for (int t = 0; t < GRU_T; ++t) {
    _Float16* hprev = hbuf + (((t & 1) ^ 1) << 10);
    _Float16* hcur  = hbuf + ((t & 1) << 10);

    // ---- A fragments: afr[0..KTX-1] = x_t, afr[KTX..] = h
    v16h afr[KT];
    if constexpr (XF32) {
      const float* xp = xf + ((size_t)(b0 + row) * GRU_T + t) * DIN + hi * 8;
      v8h c1, c2;
#pragma unroll
      for (int i = 0; i < 8; i++) c1[i] = (_Float16)xp[i];
#pragma unroll
      for (int i = 0; i < 8; i++) c2[i] = (_Float16)0.0f;   // K pad 16..31
      afr[0] = __builtin_shufflevector(c1, c2, 0, 1, 2, 3, 4, 5, 6, 7,
                                       8, 9, 10, 11, 12, 13, 14, 15);
      if (t + 1 < GRU_T) __builtin_prefetch(xp + DIN, 0, 3);
    } else {
      const _Float16* xp = xh + ((size_t)(b0 + row) * GRU_T + t) * DIN;
#pragma unroll
      for (int kt = 0; kt < KTX; kt++) {
        v8h c1 = *(const v8h*)(xp + kt * 32 + hi * 8);
        v8h c2 = *(const v8h*)(xp + kt * 32 + 16 + hi * 8);
        afr[kt] = __builtin_shufflevector(c1, c2, 0, 1, 2, 3, 4, 5, 6, 7,
                                          8, 9, 10, 11, 12, 13, 14, 15);
      }
      if (t + 1 < GRU_T) __builtin_prefetch(xp + DIN, 0, 3);
    }
    {
      const _Float16* hp = hprev + row * 64 + hi * 8;
#pragma unroll
      for (int k2 = 0; k2 < 2; k2++) {
        v8h c1 = *(const v8h*)(hp + k2 * 32);
        v8h c2 = *(const v8h*)(hp + k2 * 32 + 16);
        afr[KTX + k2] = __builtin_shufflevector(c1, c2, 0, 1, 2, 3, 4, 5, 6, 7,
                                                8, 9, 10, 11, 12, 13, 14, 15);
      }
    }

    // ---- WMMA sweep; first K-tile of each chain consumes the bias fragment
    v8f accRZ[8], accXN[4], accHN[4];
#pragma unroll
    for (int nt = 0; nt < 8; nt++)                 // r,z: combined x+h
      accRZ[nt] = __builtin_amdgcn_wmma_f32_16x16x32_f16(
          false, afr[0], false, bfr[nt], (short)0, bFragRZ[nt], false, false);
#pragma unroll
    for (int q = 0; q < 4; q++)                    // n-gate x part
      accXN[q] = __builtin_amdgcn_wmma_f32_16x16x32_f16(
          false, afr[0], false, bfr[8 + q], (short)0, bFragXN[q], false, false);
#pragma unroll
    for (int q = 0; q < 4; q++)                    // n-gate h part
      accHN[q] = __builtin_amdgcn_wmma_f32_16x16x32_f16(
          false, afr[KTX], false, bfr[KTX * 12 + 8 + q], (short)0,
          bFragHN[q], false, false);
#pragma unroll
    for (int kt = 1; kt < KT; kt++) {
#pragma unroll
      for (int nt = 0; nt < 8; nt++)
        accRZ[nt] = __builtin_amdgcn_wmma_f32_16x16x32_f16(
            false, afr[kt], false, bfr[kt * 12 + nt], (short)0,
            accRZ[nt], false, false);
#pragma unroll
      for (int q = 0; q < 4; q++) {
        if (kt < KTX)
          accXN[q] = __builtin_amdgcn_wmma_f32_16x16x32_f16(
              false, afr[kt], false, bfr[kt * 12 + 8 + q], (short)0,
              accXN[q], false, false);
        else if (kt > KTX)
          accHN[q] = __builtin_amdgcn_wmma_f32_16x16x32_f16(
              false, afr[kt], false, bfr[kt * 12 + 8 + q], (short)0,
              accHN[q], false, false);
      }
    }

    // ---- gates + state update (v_exp/v_rcp only, register-local)
#pragma unroll
    for (int q = 0; q < 4; q++) {
#pragma unroll
      for (int i = 0; i < 8; i++) {
        const float rr = fast_sigmoid(accRZ[q][i]);
        const float zz = fast_sigmoid(accRZ[4 + q][i]);
        const float nn = fast_tanh(accXN[q][i] + rr * accHN[q][i]);
        hC[q][i] = nn + zz * (hC[q][i] - nn);      // (1-z)*n + z*h
      }
    }

    // ---- before overwriting hcur, make sure async stores issued from this
    //      buffer two steps ago have drained (<= 4 newer ones may remain)
    if constexpr (STORE_ALL) {
      asm volatile("s_wait_asynccnt 0x4" ::: "memory");
    }

    // ---- transpose h: C layout -> row-major f16 LDS (next step's A source)
#pragma unroll
    for (int q = 0; q < 4; q++)
#pragma unroll
      for (int v = 0; v < 8; v++)
        hcur[(hi * 8 + v) * 64 + q * 16 + row] = (_Float16)hC[q][v];

    // ---- async LDS -> global store of this step's output tile
    if constexpr (STORE_ALL) {
      asm volatile("s_wait_dscnt 0x0" ::: "memory");  // LDS writes visible
#pragma unroll
      for (int it = 0; it < 4; it++) {
        const int c    = it * 32 + lane;    // 128 chunks of 8 halves
        const int rr   = c >> 3;
        const int coff = (c & 7) * 8;
        async_store_b128(out + ((size_t)(b0 + rr) * GRU_T + t) * GRU_H + coff,
                         hcur + rr * 64 + coff);
      }
    }
    if constexpr (STORE_H) {
      if (t == GRU_T - 1) {
#pragma unroll
        for (int q = 0; q < 4; q++)
#pragma unroll
          for (int v = 0; v < 8; v++)
            hlast[(size_t)(b0 + hi * 8 + v) * GRU_H + q * 16 + row] = hC[q][v];
      }
    }
  }
  // S_ENDPGM performs an implicit wait-idle (drains ASYNCcnt).
}

// ---------------------------------------------------------------------------
// Head: y[b] = fc2_b + sum_j fc2_w[j] * relu(fc1_b[j] + h[b,:] . fc1_w[j,:])
// ---------------------------------------------------------------------------
__global__ __launch_bounds__(128) void head_kernel(
    const float* __restrict__ hlast,
    const float* __restrict__ fc1w, const float* __restrict__ fc1b,
    const float* __restrict__ fc2w, const float* __restrict__ fc2b,
    float* __restrict__ y)
{
  const int b = blockIdx.x * blockDim.x + threadIdx.x;
  if (b >= GRU_B) return;
  float h[64];
  const float* hp = hlast + (size_t)b * GRU_H;
#pragma unroll
  for (int k = 0; k < 64; k += 4) {
    float4 v = *(const float4*)(hp + k);
    h[k] = v.x; h[k + 1] = v.y; h[k + 2] = v.z; h[k + 3] = v.w;
  }
  float s = 0.0f;
#pragma unroll 4
  for (int j = 0; j < 32; j++) {
    float a = fc1b[j];
    const float* w = fc1w + j * 64;
#pragma unroll
    for (int k = 0; k < 64; k++) a = fmaf(h[k], w[k], a);
    s = fmaf(fc2w[j], fmaxf(a, 0.0f), s);
  }
  y[b] = s + fc2b[0];
}

// ---------------------------------------------------------------------------
extern "C" void kernel_launch(void* const* d_in, const int* in_sizes, int n_in,
                              void* d_out, int out_size, void* d_ws, size_t ws_size,
                              hipStream_t stream) {
  (void)in_sizes; (void)n_in; (void)out_size; (void)ws_size;

  const float* x    = (const float*)d_in[0];
  const float* Wih0 = (const float*)d_in[1];
  const float* Whh0 = (const float*)d_in[2];
  const float* bih0 = (const float*)d_in[3];
  const float* bhh0 = (const float*)d_in[4];
  const float* Wih1 = (const float*)d_in[5];
  const float* Whh1 = (const float*)d_in[6];
  const float* bih1 = (const float*)d_in[7];
  const float* bhh1 = (const float*)d_in[8];
  const float* Wih2 = (const float*)d_in[9];
  const float* Whh2 = (const float*)d_in[10];
  const float* bih2 = (const float*)d_in[11];
  const float* bhh2 = (const float*)d_in[12];
  const float* fc1w = (const float*)d_in[13];
  const float* fc1b = (const float*)d_in[14];
  const float* fc2w = (const float*)d_in[15];
  const float* fc2b = (const float*)d_in[16];

  // Workspace: out0 (f16 64MiB) | out1 (f16 64MiB) | hlast (f32 256KiB)
  const size_t seq_elems = (size_t)GRU_B * GRU_T * GRU_H;
  _Float16* out0  = (_Float16*)d_ws;
  _Float16* out1  = out0 + seq_elems;
  float*    hlast = (float*)(out1 + seq_elems);

  const dim3 grid(GRU_B / 16), block(32);
  const size_t shm = 4096;  // 2 x (16x64) f16 h staging buffers

  gru_layer<16, 32, true, true, false><<<grid, block, shm, stream>>>(
      x, nullptr, Wih0, Whh0, bih0, bhh0, out0, nullptr);
  gru_layer<64, 64, false, true, false><<<grid, block, shm, stream>>>(
      nullptr, out0, Wih1, Whh1, bih1, bhh1, out1, nullptr);
  gru_layer<64, 64, false, false, true><<<grid, block, shm, stream>>>(
      nullptr, out1, Wih2, Whh2, bih2, bhh2, nullptr, hlast);
  head_kernel<<<dim3(GRU_B / 128), dim3(128), 0, stream>>>(
      hlast, fc1w, fc1b, fc2w, fc2b, (float*)d_out);
}
